// MinkowskiFeatureExtractor_67929202753917
// MI455X (gfx1250) — compile-verified, matrix operand order checked by reading, not run on hardware
//
#include <hip/hip_runtime.h>

#define NPTS 60000

typedef __attribute__((ext_vector_type(16))) _Float16 v16h;
typedef __attribute__((ext_vector_type(8)))  _Float16 v8h;
typedef __attribute__((ext_vector_type(8)))  float    v8f;
typedef __attribute__((ext_vector_type(4)))  unsigned int v4u;
typedef __attribute__((ext_vector_type(4)))  int v4i;
typedef __attribute__((ext_vector_type(8)))  int v8i;

union V16 { v16h v; v8h h[2]; };

static constexpr int ilog2c(int v) { int r = 0; while (v > 1) { v >>= 1; ++r; } return r; }

// ---------------------------------------------------------------------------
// Weight swizzle: f32 [Kk][Cin][Cout] -> f16 fragment layout
// dst[(((k*KK+kk)*NTILES+nt)*32+lane)*16 + j] = W[ks][kk*32+K(lane,j)][nt*16+lane%16]
// K(lane,j) = j + (j>=8?8:0) + (lane>=16?8:0); zero-pad K >= Cin; flip bakes convT.
// ---------------------------------------------------------------------------
__global__ __launch_bounds__(256)
void swizzle_weights_kernel(const float* __restrict__ src, _Float16* __restrict__ dst,
                            int Kk, int Cin, int Cout, int flip)
{
    int KK = (Cin + 31) >> 5, NTILES = Cout >> 4;
    size_t total = (size_t)Kk * KK * NTILES * 512;
    for (size_t g = (size_t)blockIdx.x * blockDim.x + threadIdx.x; g < total;
         g += (size_t)gridDim.x * blockDim.x) {
        int j    = (int)(g & 15);
        int lane = (int)((g >> 4) & 31);
        size_t r = g >> 9;
        int nt = (int)(r % NTILES); r /= NTILES;
        int kk = (int)(r % KK);
        int k  = (int)(r / KK);
        int K  = (kk << 5) + j + ((j >= 8) ? 8 : 0) + ((lane >= 16) ? 8 : 0);
        int n  = (nt << 4) + (lane & 15);
        int ks = flip ? (Kk - 1 - k) : k;
        float v = (K < Cin) ? src[((size_t)ks * Cin + K) * Cout + n] : 0.f;
        dst[g] = (_Float16)v;
    }
}

// ---------------------------------------------------------------------------
// TDM gather-mode issue: DMA 16 rows (16-bit indices in g2/g3) of a dense
// [NPTS, rowLen] f16 tensor into LDS at ldsByteOff, padding each row of
// rowLen halves with padAmount DWORDs so rows land at the WMMA row stride.
// OOB indices (0xFFFF from nbr == -1) read as zeros (the conv mask).
// ---------------------------------------------------------------------------
__device__ __forceinline__ void tdm_gather16(const _Float16* gbase, unsigned ldsByteOff,
                                             int rowLenHalves, int padIntervalCode,
                                             int padAmountCode, v4i g2, v4i g3)
{
    unsigned long long ga = (unsigned long long)(const void*)gbase;
    v4u g0;
    g0[0] = 0x80000001u;                                   // gather_mode | count=1
    g0[1] = ldsByteOff;                                    // lds_addr
    g0[2] = (unsigned)ga;                                  // global_addr lo
    g0[3] = ((unsigned)(ga >> 32) & 0x01FFFFFFu) | (2u << 30);  // addr hi | type=2
    v8i g1;
    g1[0] = (1 << 16)                                      // data_size = 2B
          | (1 << 20)                                      // pad_enable
          | (padIntervalCode << 22) | (padAmountCode << 25);
    g1[1] = (rowLenHalves & 0xFFFF) << 16;                 // tensor_dim0[15:0]
    g1[2] = (NPTS & 0xFFFF) << 16;                         // tensor_dim1[15:0]
    g1[3] = (rowLenHalves & 0xFFFF) << 16;                 // tile_dim0
    g1[4] = 16;                                            // tile_dim1 = #indices
    g1[5] = rowLenHalves;                                  // tensor_dim0_stride lo
    g1[6] = 0;
    g1[7] = 0;
#if __clang_major__ >= 23
    v8i gz = {};
    __builtin_amdgcn_tensor_load_to_lds(g0, g1, g2, g3, gz, 0);
#else
    __builtin_amdgcn_tensor_load_to_lds(g0, g1, g2, g3, 0);
#endif
}

// ---------------------------------------------------------------------------
// TDM-fed WMMA sparse conv (Cin multiple of 32, f16 activations):
//   out[i,:] = sum_k W[k] @ concat(x1h,x2h)[nbr[i,k],:]
// Wave 0 drives a double-buffered gather pipeline with TENSORcnt waits;
// compute waves consume LDS tiles with v_wmma_f32_16x16x32_f16.
// All wave-level predicates go through readfirstlane -> scalar branches,
// so EXEC stays all-ones through the WMMA bodies with no masked merges.
// ---------------------------------------------------------------------------
template<int C1, int C2, int COUT, int NT, bool SPARSE>
__global__ __launch_bounds__(256)
void conv_tdm_kernel(const _Float16* __restrict__ x1h,
                     const _Float16* __restrict__ x2h,
                     const _Float16* __restrict__ Wsw,
                     const int* __restrict__ nbr,
                     const float* __restrict__ bias, int relu,
                     float* __restrict__ out)
{
    constexpr int Cin       = C1 + C2;
    constexpr int KK        = Cin / 32;
    constexpr int NTILES    = COUT / 16;
    constexpr int rowStride = Cin + 8;          // halves
    constexpr int BUFH      = 16 * rowStride;   // halves per LDS buffer
    constexpr int Kk        = SPARSE ? 27 : 1;
    constexpr int NOPS      = (C2 > 0) ? 2 : 1; // TDM issues per k
    constexpr int IC1 = ilog2c(C1) - 2;         // pad interval: C1*2 bytes
    constexpr int PA1 = (C2 + 8) / 2 - 1;       // pad amount: (C2+8) halves
    constexpr int IC2 = (C2 > 0) ? (ilog2c(C2) - 2) : 0;
    constexpr int PA2 = (C1 + 8) / 2 - 1;

    extern __shared__ _Float16 sbuf[];          // 2 * BUFH halves

    const int tid  = threadIdx.x;
    const int lane = tid & 31;
    const int wid  = tid >> 5;
    const int i0   = blockIdx.x * 16;
    const int swid = __builtin_amdgcn_readfirstlane(wid);   // SGPR copy of wave id

    v8f acc[NT] = {};
    const int row     = lane & 15;
    const int laneoff = (lane >> 4) << 3;       // 0 or 8
    const bool computeWave = (NT == 2) || (swid < NTILES);  // scalar branch
    const bool isWave0     = (swid == 0);

    // wave-0 helper: issue the gather(s) for offset k into LDS buffer bi
    auto issue = [&](int k, int bi) {
        int v = -1;
        if (lane < 16)
            v = SPARSE ? nbr[(i0 + lane) * 27 + k] : (i0 + lane);
        unsigned u[16];
#pragma unroll
        for (int m = 0; m < 16; ++m)
            u[m] = (unsigned)__builtin_amdgcn_readlane(v, m) & 0xFFFFu;
        v4i g2, g3;
        g2[0] = (int)(u[0]  | (u[1]  << 16));
        g2[1] = (int)(u[2]  | (u[3]  << 16));
        g2[2] = (int)(u[4]  | (u[5]  << 16));
        g2[3] = (int)(u[6]  | (u[7]  << 16));
        g3[0] = (int)(u[8]  | (u[9]  << 16));
        g3[1] = (int)(u[10] | (u[11] << 16));
        g3[2] = (int)(u[12] | (u[13] << 16));
        g3[3] = (int)(u[14] | (u[15] << 16));
        unsigned ldsoff = (unsigned)(bi * BUFH * 2);
        tdm_gather16(x1h, ldsoff, C1, IC1, PA1, g2, g3);
        if (C2 > 0)
            tdm_gather16(x2h, ldsoff + C1 * 2, C2, IC2, PA2, g2, g3);
    };

    if (isWave0) issue(0, 0);

    for (int k = 0; k < Kk; ++k) {
        if (isWave0) {
            if (k + 1 < Kk) {
                issue(k + 1, (k + 1) & 1);
                __builtin_amdgcn_s_wait_tensorcnt(NOPS);   // buffer k complete
            } else {
                __builtin_amdgcn_s_wait_tensorcnt(0);
            }
        }
        __syncthreads();                        // publish buffer k to all waves

        if (computeWave) {
            const _Float16* buf = sbuf + (k & 1) * BUFH;
            const _Float16* wk  = Wsw + (size_t)k * KK * NTILES * 512;
#pragma unroll
            for (int kk = 0; kk < KK; ++kk) {
                V16 a;
                const _Float16* ap = &buf[row * rowStride + (kk << 5) + laneoff];
                a.h[0] = *(const v8h*)(ap);     // halves j=0..7  -> K=base+0..7
                a.h[1] = *(const v8h*)(ap + 16);// halves j=8..15 -> K=base+16..23
#pragma unroll
                for (int t = 0; t < NT; ++t) {
                    int nt = wid + t * 8;       // < NTILES by construction
                    V16 b;
                    const _Float16* bp =
                        wk + (((size_t)(kk * NTILES + nt) << 5) + lane) * 16;
                    b.h[0] = *(const v8h*)(bp);
                    b.h[1] = *(const v8h*)(bp + 8);
                    acc[t] = __builtin_amdgcn_wmma_f32_16x16x32_f16(
                        false, a.v, false, b.v, (short)0, acc[t], false, false);
                }
            }
        }
        __syncthreads();                        // done reading before reuse
    }

    // C/D layout: VGPR r -> M = r + (lane>=16 ? 8 : 0), N = lane&15
    if (computeWave) {
#pragma unroll
        for (int t = 0; t < NT; ++t) {
            int nt = wid + t * 8;
            int n = (nt << 4) + (lane & 15);
            int mbase = (lane >> 4) << 3;
            float bv = bias ? bias[n] : 0.f;
#pragma unroll
            for (int r = 0; r < 8; ++r) {
                float v = acc[t][r] + bv;
                if (relu) v = fmaxf(v, 0.f);
                out[(size_t)(i0 + mbase + r) * COUT + n] = v;
            }
        }
    }
}

// ---------------------------------------------------------------------------
// VALU-gather WMMA conv (f32 input) — kept only for the 3-channel stem.
// ---------------------------------------------------------------------------
template<int CIN, int COUT>
__global__ __launch_bounds__(256)
void conv_gather_kernel(const float* __restrict__ x1,
                        const _Float16* __restrict__ Wsw,
                        const int* __restrict__ nbr,
                        float* __restrict__ out)
{
    constexpr int Cin_pad   = (CIN + 31) & ~31;
    constexpr int KK        = Cin_pad / 32;
    constexpr int NTILES    = COUT / 16;
    constexpr int rowStride = Cin_pad + 8;

    __shared__ _Float16 sA[16 * rowStride];
    __shared__ int sIdx[16];

    const int tid  = threadIdx.x;
    const int lane = tid & 31;
    const int wid  = tid >> 5;
    const int i0   = blockIdx.x * 16;
    const int swid = __builtin_amdgcn_readfirstlane(wid);

    if constexpr (Cin_pad != CIN) {
        constexpr int padw = Cin_pad - CIN;
        for (int e = tid; e < 16 * padw; e += 256) {
            int m = e / padw, col = CIN + (e - m * padw);
            sA[m * rowStride + col] = (_Float16)0.f;
        }
    }

    v8f acc = {};
    const int row     = lane & 15;
    const int laneoff = (lane >> 4) << 3;
    const bool computeWave = (swid < NTILES);   // scalar branch

    for (int k = 0; k < 27; ++k) {
        __syncthreads();
        if (tid < 16) sIdx[tid] = nbr[(i0 + tid) * 27 + k];
        __syncthreads();
        for (int e = tid; e < 16 * CIN; e += 256) {
            int m = e / CIN, col = e - m * CIN;
            int j  = sIdx[m];
            int jc = (j < 0) ? 0 : j;                      // branchless mask
            float v = x1[(size_t)jc * CIN + col];
            v = (j < 0) ? 0.f : v;
            sA[m * rowStride + col] = (_Float16)v;
        }
        __syncthreads();

        if (computeWave) {
            const _Float16* wk = Wsw + (size_t)k * KK * NTILES * 512;
#pragma unroll
            for (int kk = 0; kk < KK; ++kk) {
                V16 a;
                const _Float16* ap = &sA[row * rowStride + (kk << 5) + laneoff];
                a.h[0] = *(const v8h*)(ap);
                a.h[1] = *(const v8h*)(ap + 16);
                V16 b;
                const _Float16* bp =
                    wk + (((size_t)(kk * NTILES + wid) << 5) + lane) * 16;
                b.h[0] = *(const v8h*)(bp);
                b.h[1] = *(const v8h*)(bp + 8);
                acc = __builtin_amdgcn_wmma_f32_16x16x32_f16(
                    false, a.v, false, b.v, (short)0, acc, false, false);
            }
        }
    }

    if (computeWave) {
        int n = (wid << 4) + (lane & 15);
        int mbase = (lane >> 4) << 3;
#pragma unroll
        for (int r = 0; r < 8; ++r)
            out[(size_t)(i0 + mbase + r) * COUT + n] = acc[r];
    }
}

// ---------------------------------------------------------------------------
// Per-channel mean / rstd over all N points
// ---------------------------------------------------------------------------
__global__ __launch_bounds__(256)
void colstats_kernel(const float* __restrict__ x, int n, int C,
                     float* __restrict__ mean, float* __restrict__ rstd)
{
    __shared__ float sh[256], sh2[256];
    int c = blockIdx.x;
    float s = 0.f, s2 = 0.f;
    for (int i = threadIdx.x; i < n; i += 256) {
        float v = x[(size_t)i * C + c];
        s += v; s2 += v * v;
    }
    sh[threadIdx.x] = s; sh2[threadIdx.x] = s2;
    __syncthreads();
    for (int o = 128; o > 0; o >>= 1) {
        if (threadIdx.x < o) {
            sh[threadIdx.x]  += sh[threadIdx.x + o];
            sh2[threadIdx.x] += sh2[threadIdx.x + o];
        }
        __syncthreads();
    }
    if (threadIdx.x == 0) {
        float m   = sh[0] / (float)n;
        float var = sh2[0] / (float)n - m * m;
        mean[c] = m;
        rstd[c] = rsqrtf(var + 1e-5f);
    }
}

// ---------------------------------------------------------------------------
// Fused norm-apply: out = [relu]( (x-mean)*rstd [*g] [+b] [+res] )
// Also writes the f16 activation shadow (outh) consumed by the TDM convs.
// ---------------------------------------------------------------------------
__global__ __launch_bounds__(256)
void apply_norm_kernel(const float* __restrict__ x, const float* __restrict__ mean,
                       const float* __restrict__ rstd, const float* __restrict__ g,
                       const float* __restrict__ b, const float* __restrict__ res,
                       int relu, float* __restrict__ out, _Float16* __restrict__ outh,
                       int n, int Cmask)
{
    size_t total = (size_t)n * (Cmask + 1);
    for (size_t i = (size_t)blockIdx.x * blockDim.x + threadIdx.x; i < total;
         i += (size_t)gridDim.x * blockDim.x) {
        int c = (int)(i & (size_t)Cmask);
        float v = (x[i] - mean[c]) * rstd[c];
        if (g) v *= g[c];
        if (b) v += b[c];
        if (res) v += res[i];
        if (relu) v = fmaxf(v, 0.f);
        out[i] = v;
        if (outh) outh[i] = (_Float16)v;
    }
}

__global__ __launch_bounds__(256)
void f32_to_f16_kernel(const float* __restrict__ x, _Float16* __restrict__ y, size_t n)
{
    for (size_t i = (size_t)blockIdx.x * blockDim.x + threadIdx.x; i < n;
         i += (size_t)gridDim.x * blockDim.x)
        y[i] = (_Float16)x[i];
}

// ---------------------------------------------------------------------------
// Host-side helpers
// ---------------------------------------------------------------------------
static inline size_t sws(int Kk, int Cin, int Cout)
{
    return (size_t)Kk * ((Cin + 31) / 32) * (Cout / 16) * 512;
}

template<int C1, int C2, int COUT, bool SPARSE>
static void conv_tdm_l(hipStream_t s, const _Float16* x1h, const _Float16* x2h,
                       const _Float16* w, const int* nbr,
                       const float* bias, int relu, float* outp)
{
    constexpr int NT = (COUT / 16 > 8) ? 2 : 1;
    constexpr size_t shbytes = 2u * 16u * (C1 + C2 + 8) * 2u;
    conv_tdm_kernel<C1, C2, COUT, NT, SPARSE>
        <<<NPTS / 16, 256, shbytes, s>>>(x1h, x2h, w, nbr, bias, relu, outp);
}

static void norm_op(hipStream_t s, const float* x, int C, const float* g,
                    const float* b, const float* res, int relu, float* outp,
                    _Float16* outh, float* mean, float* rstd)
{
    colstats_kernel<<<C, 256, 0, s>>>(x, NPTS, C, mean, rstd);
    size_t total = (size_t)NPTS * C;
    int blocks = (int)((total + 255) / 256);
    if (blocks > 4096) blocks = 4096;
    apply_norm_kernel<<<blocks, 256, 0, s>>>(x, mean, rstd, g, b, res, relu, outp,
                                             outh, NPTS, C - 1);
}

// BN res-block tail: two convs + two bnorm applies, residual in sout/south
template<int COUT>
static void block_bn(hipStream_t s, const _Float16* w1, const _Float16* w2,
                     const float* g1, const float* b1,
                     const float* g2, const float* b2, const int* nbr,
                     float* Ta, float* Tb, _Float16* Tbh,
                     float* mean, float* rstd, float* sout, _Float16* south)
{
    conv_tdm_l<COUT, 0, COUT, true>(s, south, nullptr, w1, nbr, nullptr, 0, Ta);
    norm_op(s, Ta, COUT, g1, b1, nullptr, 1, Tb, Tbh, mean, rstd);
    conv_tdm_l<COUT, 0, COUT, true>(s, Tbh, nullptr, w2, nbr, nullptr, 0, Ta);
    norm_op(s, Ta, COUT, g2, b2, sout, 1, sout, south, mean, rstd);
}

// decoder IN res-block on Dd/Ddh (C channels), in place
template<int C>
static void block_in(hipStream_t s, float* Dd, _Float16* Ddh,
                     const _Float16* w1, const _Float16* w2, const int* nbr,
                     float* Ta, float* Tb, _Float16* Tbh,
                     float* mean, float* rstd)
{
    conv_tdm_l<C, 0, C, true>(s, Ddh, nullptr, w1, nbr, nullptr, 0, Ta);
    norm_op(s, Ta, C, nullptr, nullptr, nullptr, 1, Tb, Tbh, mean, rstd);
    conv_tdm_l<C, 0, C, true>(s, Tbh, nullptr, w2, nbr, nullptr, 0, Ta);
    norm_op(s, Ta, C, nullptr, nullptr, Dd, 1, Dd, Ddh, mean, rstd);
}

extern "C" void kernel_launch(void* const* d_in, const int* in_sizes, int n_in,
                              void* d_out, int out_size, void* d_ws, size_t ws_size,
                              hipStream_t stream)
{
    (void)in_sizes; (void)n_in; (void)out_size; (void)ws_size;
#define PF(i) ((const float*)d_in[(i)])

    const float* feats[2] = { PF(0), PF(1) };
    const int*   nbrs[2]  = { (const int*)d_in[48], (const int*)d_in[49] };

    char* ws = (char*)d_ws;
    size_t off = 0;
    auto halloc = [&](size_t nh) -> _Float16* {
        _Float16* p = (_Float16*)(ws + off);
        off += ((nh * 2 + 255) / 256) * 256;
        return p;
    };
    auto falloc = [&](size_t nf) -> float* {
        float* p = (float*)(ws + off);
        off += ((nf * 4 + 255) / 256) * 256;
        return p;
    };

    auto swz = [&](const float* src, int Kk, int Cin, int Cout, int flip) -> const _Float16* {
        size_t nh = sws(Kk, Cin, Cout);
        _Float16* dst = halloc(nh);
        int blocks = (int)((nh + 255) / 256);
        if (blocks > 8192) blocks = 8192;
        swizzle_weights_kernel<<<blocks, 256, 0, stream>>>(src, dst, Kk, Cin, Cout, flip);
        return dst;
    };

    // params flattened in setup_inputs() insertion order (indices 2..47)
    const _Float16* w_ec1  = swz(PF(2),  27,   3,  32, 0);
    const _Float16* w_eb1a = swz(PF(3),  27,  32,  32, 0);
    const _Float16* w_eb1b = swz(PF(6),  27,  32,  32, 0);
    const _Float16* w_ec2  = swz(PF(9),  27,  32,  64, 0);
    const _Float16* w_eb2a = swz(PF(10), 27,  64,  64, 0);
    const _Float16* w_eb2b = swz(PF(13), 27,  64,  64, 0);
    const _Float16* w_ec3  = swz(PF(16), 27,  64, 128, 0);
    const _Float16* w_eb3a = swz(PF(17), 27, 128, 128, 0);
    const _Float16* w_eb3b = swz(PF(20), 27, 128, 128, 0);
    const _Float16* w_ec4  = swz(PF(23), 27, 128, 256, 0);
    const _Float16* w_eb4a = swz(PF(24), 27, 256, 256, 0);
    const _Float16* w_eb4b = swz(PF(27), 27, 256, 256, 0);
    const _Float16* w_dc4  = swz(PF(30), 27, 256, 128, 1);   // transpose conv: flip
    const _Float16* w_bl4a = swz(PF(33), 27, 128, 128, 0);
    const _Float16* w_bl4b = swz(PF(34), 27, 128, 128, 0);
    const _Float16* w_dc3  = swz(PF(35), 27, 256,  64, 1);
    const _Float16* w_bl3a = swz(PF(38), 27,  64,  64, 0);
    const _Float16* w_bl3b = swz(PF(39), 27,  64,  64, 0);
    const _Float16* w_dc2  = swz(PF(40), 27, 128,  64, 1);
    const _Float16* w_bl2a = swz(PF(43), 27,  64,  64, 0);
    const _Float16* w_bl2b = swz(PF(44), 27,  64,  64, 0);
    const _Float16* w_dc1  = swz(PF(45),  1,  96,  32, 0);   // 1x1 head
    const _Float16* w_fin  = swz(PF(46),  1,  32, 128, 0);

    float* Ta = falloc((size_t)NPTS * 256);
    float* Tb = falloc((size_t)NPTS * 256);
    float* s1 = falloc((size_t)NPTS * 32);
    float* s2 = falloc((size_t)NPTS * 64);
    float* s4 = falloc((size_t)NPTS * 128);
    float* s8 = falloc((size_t)NPTS * 256);
    float* Dd = falloc((size_t)NPTS * 128);
    float* mean = falloc(256);
    float* rstd = falloc(256);
    _Float16* s1h = halloc((size_t)NPTS * 32);
    _Float16* s2h = halloc((size_t)NPTS * 64);
    _Float16* s4h = halloc((size_t)NPTS * 128);
    _Float16* s8h = halloc((size_t)NPTS * 256);
    _Float16* Tbh = halloc((size_t)NPTS * 256);
    _Float16* Ddh = halloc((size_t)NPTS * 128);

    for (int cl = 0; cl < 2; ++cl) {
        const int* nbr = nbrs[cl];
        float* outp = (float*)d_out + (size_t)cl * NPTS * 128;

        // encoder stage 1 (f32 stem via VALU gather), then TDM everywhere
        conv_gather_kernel<3, 32><<<NPTS / 16, 256, 0, stream>>>(feats[cl], w_ec1, nbr, Ta);
        norm_op(stream, Ta, 32, nullptr, nullptr, nullptr, 0, s1, s1h, mean, rstd);
        block_bn<32>(stream, w_eb1a, w_eb1b, PF(4), PF(5), PF(7), PF(8), nbr,
                     Ta, Tb, Tbh, mean, rstd, s1, s1h);

        conv_tdm_l<32, 0, 64, true>(stream, s1h, nullptr, w_ec2, nbr, nullptr, 0, Ta);
        norm_op(stream, Ta, 64, nullptr, nullptr, nullptr, 0, s2, s2h, mean, rstd);
        block_bn<64>(stream, w_eb2a, w_eb2b, PF(11), PF(12), PF(14), PF(15), nbr,
                     Ta, Tb, Tbh, mean, rstd, s2, s2h);

        conv_tdm_l<64, 0, 128, true>(stream, s2h, nullptr, w_ec3, nbr, nullptr, 0, Ta);
        norm_op(stream, Ta, 128, nullptr, nullptr, nullptr, 0, s4, s4h, mean, rstd);
        block_bn<128>(stream, w_eb3a, w_eb3b, PF(18), PF(19), PF(21), PF(22), nbr,
                      Ta, Tb, Tbh, mean, rstd, s4, s4h);

        conv_tdm_l<128, 0, 256, true>(stream, s4h, nullptr, w_ec4, nbr, nullptr, 0, Ta);
        norm_op(stream, Ta, 256, nullptr, nullptr, nullptr, 0, s8, s8h, mean, rstd);
        block_bn<256>(stream, w_eb4a, w_eb4b, PF(25), PF(26), PF(28), PF(29), nbr,
                      Ta, Tb, Tbh, mean, rstd, s8, s8h);

        // d4: o = block_in(bnorm(convT(s8)))
        conv_tdm_l<256, 0, 128, true>(stream, s8h, nullptr, w_dc4, nbr, nullptr, 0, Ta);
        norm_op(stream, Ta, 128, PF(31), PF(32), nullptr, 0, Dd, Ddh, mean, rstd);
        block_in<128>(stream, Dd, Ddh, w_bl4a, w_bl4b, nbr, Ta, Tb, Tbh, mean, rstd);
        // d3: concat(o, s4) fused into the TDM gather (two descriptors per k)
        conv_tdm_l<128, 128, 64, true>(stream, Ddh, s4h, w_dc3, nbr, nullptr, 0, Ta);
        norm_op(stream, Ta, 64, PF(36), PF(37), nullptr, 0, Dd, Ddh, mean, rstd);
        block_in<64>(stream, Dd, Ddh, w_bl3a, w_bl3b, nbr, Ta, Tb, Tbh, mean, rstd);
        // d2: concat(o, s2)
        conv_tdm_l<64, 64, 64, true>(stream, Ddh, s2h, w_dc2, nbr, nullptr, 0, Ta);
        norm_op(stream, Ta, 64, PF(41), PF(42), nullptr, 0, Dd, Ddh, mean, rstd);
        block_in<64>(stream, Dd, Ddh, w_bl2a, w_bl2b, nbr, Ta, Tb, Tbh, mean, rstd);
        // head: relu(concat(o, s1) @ d_c1) then @ fin_W + fin_b
        conv_tdm_l<64, 32, 32, false>(stream, Ddh, s1h, w_dc1, nullptr, nullptr, 1, Tb);
        {
            size_t n = (size_t)NPTS * 32;
            int blocks = (int)((n + 255) / 256);
            if (blocks > 4096) blocks = 4096;
            f32_to_f16_kernel<<<blocks, 256, 0, stream>>>(Tb, Tbh, n);
        }
        conv_tdm_l<32, 0, 128, false>(stream, Tbh, nullptr, w_fin, nullptr, PF(47), 0, outp);
    }
#undef PF
}
